// Sp_GCN_89799176225667
// MI455X (gfx1250) — matrix operand order repaired and checked by reading, not compiled, not run on hardware
//
#include <hip/hip_runtime.h>
#include <hip/hip_bf16.h>

typedef __attribute__((ext_vector_type(16))) __bf16 v16bf;
typedef __attribute__((ext_vector_type(8)))  __bf16 v8bf;
typedef __attribute__((ext_vector_type(8)))  float  v8f;

#define GCN_N_NODES 100000
#define GCN_F_IN    256
#define GCN_H       128
#define GCN_N_EDGES 3200000

// ---------------------------------------------------------------------------
// CDNA5 async copy helpers: GLOBAL_LOAD_ASYNC_TO_LDS_* (tracked by ASYNCcnt).
// LDS destination address is passed as an addrspace(3) pointer (one VGPR).
// ---------------------------------------------------------------------------
__device__ __forceinline__ void async_copy_b128(void* lds_dst, const void* gsrc)
{
    auto l = (__attribute__((address_space(3))) char*)lds_dst;
    asm volatile("global_load_async_to_lds_b128 %0, %1, off"
                 :: "v"(l), "v"(gsrc) : "memory");
}
__device__ __forceinline__ void wait_async0()
{
    asm volatile("s_wait_asynccnt 0x0" ::: "memory");
}

// ---------------------------------------------------------------------------
// One-time weight prep: Wt[n][k] = (bf16)W[k][n]  (transposed so GEMM B-tiles
// are contiguous 16B chunks for async-to-LDS staging).
// ---------------------------------------------------------------------------
__global__ __launch_bounds__(256) void gcn_convert_w(
    const float* __restrict__ W, __bf16* __restrict__ Wt, int K)
{
    int i = blockIdx.x * 256 + threadIdx.x;           // i = k*128 + n (coalesced)
    if (i < K * 128) {
        int k = i >> 7, n = i & 127;
        Wt[(long)n * K + k] = (__bf16)W[i];
    }
}

// ---------------------------------------------------------------------------
// GEMM: C[M x 128] = A[M x KDIM] @ W[KDIM x 128], bf16 WMMA, f32 accumulate.
// Block = 256 threads = 8 waves; 128x128 output tile per block.
// A (f32) and Wt (bf16, pre-transposed) staged via async-to-LDS; f32->bf16
// conversion (+ optional fused ReLU) happens in the per-wave fragment build.
// ---------------------------------------------------------------------------
template <int KDIM, bool RELU_A>
__global__ __launch_bounds__(256) void gcn_gemm_wmma(
    const float* __restrict__ A, const __bf16* __restrict__ Wt,
    float* __restrict__ C, int M)
{
    __shared__ float  At[128][36];   // [m][k] f32, 144B row (16B-aligned)
    __shared__ __bf16 Ws[128][40];   // [n][k] bf16, 80B row (16B-aligned)

    const int tid  = threadIdx.x;
    const int wave = tid >> 5;
    const int lane = tid & 31;
    const int half = lane >> 4;      // 0: lanes 0-15, 1: lanes 16-31
    const int l16  = lane & 15;
    const int mblk = blockIdx.x * 128;
    const int mrow = wave * 16;

    v8f acc[8] = {};

    for (int kk = 0; kk < KDIM; kk += 32) {
        // Stage A[mblk..+127][kk..+31] f32 as 1024 x 16B async chunks.
        #pragma unroll
        for (int ch = tid; ch < 1024; ch += 256) {
            int r = ch >> 3, c = ch & 7;             // 8 chunks (32 floats)/row
            int gr = mblk + r; if (gr >= M) gr = M - 1;
            async_copy_b128(&At[r][c * 4], A + (long)gr * KDIM + kk + c * 4);
        }
        // Stage Wt[0..127][kk..+31] bf16 as 512 x 16B async chunks.
        #pragma unroll
        for (int ch = tid; ch < 512; ch += 256) {
            int n = ch >> 2, p = ch & 3;             // 4 chunks (32 bf16)/row
            async_copy_b128(&Ws[n][p * 8], Wt + (long)n * KDIM + kk + p * 8);
        }
        wait_async0();
        __syncthreads();

        // A fragment (16x32 bf16): lanes 0-15: M=l16, K={0..7,16..23};
        //                          lanes 16-31: M=l16, K={8..15,24..31}
        const int ar = mrow + l16;
        const int k0 = half * 8;
        const float4* ap = (const float4*)&At[ar][k0];        // 32B, 16B-aligned
        const float4* aq = (const float4*)&At[ar][k0 + 16];
        float4 x0 = ap[0], x1 = ap[1], y0 = aq[0], y1 = aq[1];
        v16bf afrag;
        {
            float fa[16] = { x0.x,x0.y,x0.z,x0.w, x1.x,x1.y,x1.z,x1.w,
                             y0.x,y0.y,y0.z,y0.w, y1.x,y1.y,y1.z,y1.w };
            #pragma unroll
            for (int j = 0; j < 16; ++j) {
                float f = fa[j];
                if (RELU_A) f = fmaxf(f, 0.0f);
                afrag[j] = (__bf16)f;
            }
        }

        #pragma unroll
        for (int nt = 0; nt < 8; ++nt) {
            // B fragment (32x16 bf16): lanes 0-15: N=l16, K={0..7,16..23};
            //                          lanes 16-31: N=l16, K={8..15,24..31}
            const int nn = nt * 16 + l16;
            const int kb = half * 8;
            const v8bf b0 = *(const v8bf*)&Ws[nn][kb];
            const v8bf b1 = *(const v8bf*)&Ws[nn][kb + 16];
            const v16bf bfrag = __builtin_shufflevector(
                b0, b1, 0,1,2,3,4,5,6,7,8,9,10,11,12,13,14,15);
            acc[nt] = __builtin_amdgcn_wmma_f32_16x16x32_bf16(
                false, afrag, false, bfrag, (short)0, acc[nt], false, false);
        }
        __syncthreads();
    }

    // C/D layout: VGPR v, lanes 0-15 -> row v, lanes 16-31 -> row v+8
    const int row0 = mblk + mrow + half * 8;
    #pragma unroll
    for (int nt = 0; nt < 8; ++nt) {
        const int col = nt * 16 + l16;
        #pragma unroll
        for (int v = 0; v < 8; ++v) {
            int r = row0 + v;
            if (r < M) C[(long)r * 128 + col] = acc[nt][v];
        }
    }
}

// ---------------------------------------------------------------------------
// COO SpMM with sorted rows: Y[r] += val[e] * X[col[e]], X/Y are [N x 128].
// One wave = 256 edges; 32 lanes cover 128 features as float4. Edge metadata
// staged via async-to-LDS; gather loop software-pipelined one edge ahead.
// Register run-accumulation, f32 HW atomics only at run boundaries.
// ---------------------------------------------------------------------------
__global__ __launch_bounds__(256) void gcn_spmm_coo(
    const int* __restrict__ erow, const int* __restrict__ ecol,
    const float* __restrict__ eval, const float* __restrict__ X,
    float* __restrict__ Y, int nE)
{
    const int EPW = 256;
    __shared__ int   s_row[8][EPW];
    __shared__ int   s_col[8][EPW];
    __shared__ float s_val[8][EPW];

    const int  wave = threadIdx.x >> 5;
    const int  lane = threadIdx.x & 31;
    const long base = ((long)blockIdx.x * 8 + wave) * EPW;

    int count = 0;
    if (base < (long)nE) {
        long rem = (long)nE - base;
        count = rem < (long)EPW ? (int)rem : EPW;
    }
    // Stage 256 edges of metadata: 64 x 16B chunks per array, 2 per lane.
    #pragma unroll
    for (int ch = lane; ch < 64; ch += 32) {
        long g = base + ch * 4;                       // 4 edges per chunk
        if (g > (long)nE - 4) g = (long)nE - 4;       // clamp to tail (nE >= 4)
        async_copy_b128(&s_row[wave][ch * 4], erow + g);
        async_copy_b128(&s_col[wave][ch * 4], ecol + g);
        async_copy_b128(&s_val[wave][ch * 4], eval + g);
    }
    wait_async0();
    __syncthreads();

    if (count > 0) {
        float ax = 0.f, ay = 0.f, az = 0.f, aw = 0.f;
        int curRow = s_row[wave][0];
        float4 xv = *((const float4*)(X + (long)s_col[wave][0] * 128) + lane);
        for (int e = 0; e < count; ++e) {
            // Prefetch next edge's feature row (clamped; overlapped with FMAs)
            int en = e + 1 < count ? e + 1 : e;
            float4 nxt = *((const float4*)(X + (long)s_col[wave][en] * 128) + lane);
            const int r = s_row[wave][e];
            if (r != curRow) {                        // wave-uniform branch
                float* yp = Y + (long)curRow * 128 + lane * 4;
                unsafeAtomicAdd(yp + 0, ax); unsafeAtomicAdd(yp + 1, ay);
                unsafeAtomicAdd(yp + 2, az); unsafeAtomicAdd(yp + 3, aw);
                ax = ay = az = aw = 0.f;
                curRow = r;
            }
            const float v = s_val[wave][e];
            ax += v * xv.x; ay += v * xv.y; az += v * xv.z; aw += v * xv.w;
            xv = nxt;
        }
        float* yp = Y + (long)curRow * 128 + lane * 4;
        unsafeAtomicAdd(yp + 0, ax); unsafeAtomicAdd(yp + 1, ay);
        unsafeAtomicAdd(yp + 2, az); unsafeAtomicAdd(yp + 3, aw);
    }
}

// In-place ReLU over float4s.
__global__ __launch_bounds__(256) void gcn_relu_f4(float4* __restrict__ p, long n4)
{
    long i = (long)blockIdx.x * blockDim.x + threadIdx.x;
    const long stride = (long)gridDim.x * blockDim.x;
    for (; i < n4; i += stride) {
        float4 v = p[i];
        v.x = fmaxf(v.x, 0.f); v.y = fmaxf(v.y, 0.f);
        v.z = fmaxf(v.z, 0.f); v.w = fmaxf(v.w, 0.f);
        p[i] = v;
    }
}

extern "C" void kernel_launch(void* const* d_in, const int* in_sizes, int n_in,
                              void* d_out, int out_size, void* d_ws, size_t ws_size,
                              hipStream_t stream)
{
    const float* X    = (const float*)d_in[0];   // [100000 x 256]
    const int*   erow = (const int*)d_in[1];     // [3.2M] sorted
    const int*   ecol = (const int*)d_in[2];     // [3.2M]
    const float* eval = (const float*)d_in[3];   // [3.2M]
    const float* w0   = (const float*)d_in[4];   // [256 x 128]
    const float* w1   = (const float*)d_in[5];   // [128 x 128]
    float*       out  = (float*)d_out;           // [100000 x 128]

    const size_t nmH = (size_t)GCN_N_NODES * GCN_H;
    float*  buf0 = (float*)d_ws;                         // XW0, later H1@W1
    float*  buf1 = buf0 + nmH;                           // H1 pre-relu accum
    __bf16* wt0  = (__bf16*)(buf1 + nmH);                // [128][256] bf16
    __bf16* wt1  = wt0 + (size_t)128 * GCN_F_IN;         // [128][128] bf16

    // Zero SpMM accumulators every call (harness poisons buffers).
    hipMemsetAsync(buf1, 0, nmH * sizeof(float), stream);
    hipMemsetAsync(out,  0, nmH * sizeof(float), stream);

    // Weight prep (tiny).
    gcn_convert_w<<<(GCN_F_IN * 128 + 255) / 256, 256, 0, stream>>>(w0, wt0, GCN_F_IN);
    gcn_convert_w<<<(GCN_H   * 128 + 255) / 256, 256, 0, stream>>>(w1, wt1, GCN_H);

    const int mBlocks = (GCN_N_NODES + 127) / 128;                 // 782
    const int eBlocks = (GCN_N_EDGES + 8 * 256 - 1) / (8 * 256);   // 1563

    // Layer 1: buf0 = X @ W0 ; buf1 = Ahat @ buf0   (relu deferred into GEMM2)
    gcn_gemm_wmma<GCN_F_IN, false><<<mBlocks, 256, 0, stream>>>(X, wt0, buf0, GCN_N_NODES);
    gcn_spmm_coo<<<eBlocks, 256, 0, stream>>>(erow, ecol, eval, buf0, buf1, GCN_N_EDGES);

    // Layer 2: buf0 = relu(buf1) @ W1 ; out = Ahat @ buf0 ; out = relu(out)
    gcn_gemm_wmma<GCN_H, true><<<mBlocks, 256, 0, stream>>>(buf1, wt1, buf0, GCN_N_NODES);
    gcn_spmm_coo<<<eBlocks, 256, 0, stream>>>(erow, ecol, eval, buf0, out, GCN_N_EDGES);
    gcn_relu_f4<<<1024, 256, 0, stream>>>((float4*)out, (long)(nmH / 4));
}
